// MultiHeadAttention_45217415692658
// MI455X (gfx1250) — matrix operand order repaired and checked by reading, not compiled, not run on hardware
//
#include <hip/hip_runtime.h>
#include <stdint.h>
#include <stddef.h>

#define NH   16
#define SEQ  65
#define DK   64
#define DM   1024
#define BATCH 256
#define ROWS (BATCH * SEQ)          // 16640 = 130 * 128
#define OUT_ELEMS ((size_t)ROWS * DM)               // 17,039,360
#define ATT_ELEMS ((size_t)BATCH * NH * SEQ * SEQ)  // 17,305,600

typedef __bf16 bf16;
typedef bf16  v16bf __attribute__((ext_vector_type(16)));
typedef float v8f   __attribute__((ext_vector_type(8)));

union Frag16 { uint32_t u[8]; v16bf v; };

__device__ __forceinline__ uint16_t f2bf(float f) {
  uint32_t u = __float_as_uint(f);
  u += 0x7FFFu + ((u >> 16) & 1u);     // round-to-nearest-even
  return (uint16_t)(u >> 16);
}
__device__ __forceinline__ float bf2f(uint32_t lo16) {
  return __uint_as_float(lo16 << 16);
}
__device__ __forceinline__ uint32_t pk2(float a, float b) {
  return (uint32_t)f2bf(a) | ((uint32_t)f2bf(b) << 16);
}
__device__ __forceinline__ v8f zero8() {
  v8f z;
  #pragma unroll
  for (int i = 0; i < 8; ++i) z[i] = 0.f;
  return z;
}
__device__ __forceinline__ v8f wmma_bf16(const Frag16& a, const Frag16& b, v8f c) {
  return __builtin_amdgcn_wmma_f32_16x16x32_bf16(false, a.v, false, b.v,
                                                 (short)0, c, false, false);
}

// Pin the GEMM inner-loop issue pattern: 8 x (2 ds_reads -> 1 WMMA).
// Bounds live B-fragments to ~2 so accumulators stay pinned in the WMMA
// C/D register ranges (no v_mov shuffles, no WMMA->VALU hazard NOPs).
// Masks: 0x100 = DS read, 0x008 = MFMA/WMMA.
__device__ __forceinline__ void gemm_sched_hint() {
#if __has_builtin(__builtin_amdgcn_sched_group_barrier)
  #pragma unroll
  for (int i = 0; i < 8; ++i) {
    __builtin_amdgcn_sched_group_barrier(0x100, 2, 0);  // 2 LDS reads
    __builtin_amdgcn_sched_group_barrier(0x008, 1, 0);  // 1 WMMA
  }
#endif
}

// ---- CDNA5 async global->LDS copy (ASYNCcnt-tracked), guarded fallback ----
#if defined(__gfx1250__) && __has_builtin(__builtin_amdgcn_global_load_async_to_lds_b128) && __has_builtin(__builtin_amdgcn_s_wait_asynccnt)
#define USE_ASYNC_LDS 1
typedef int v4i __attribute__((ext_vector_type(4)));
typedef __attribute__((address_space(1))) v4i* g4p;   // global int4*
typedef __attribute__((address_space(3))) v4i* l4p;   // LDS int4*
__device__ __forceinline__ void async_g2l_b128(const void* g, void* l) {
  g4p gp = (g4p)(uintptr_t)g;                      // global addr = full 64-bit
  l4p lp = (l4p)(uint32_t)(uintptr_t)l;            // LDS addr = low 32 bits
  __builtin_amdgcn_global_load_async_to_lds_b128(gp, lp, 0, 0);
}
#endif

// ---------------------------------------------------------------- convert
// Vectorized fp32 -> bf16 (RNE), n4 = number of float4 groups.
__global__ void __launch_bounds__(256) cvt4_kernel(
    const float* __restrict__ src, uint16_t* __restrict__ dst, int n4) {
  int i = blockIdx.x * 256 + threadIdx.x;
  if (i >= n4) return;
  float4 f = ((const float4*)src)[i];
  uint2 o; o.x = pk2(f.x, f.y); o.y = pk2(f.z, f.w);
  ((uint2*)dst)[i] = o;
}

// ------------------------------------------------------- build pos tables
// pos[q][k][:] : q<64,k<64 -> fmap[(yq-yk+7)][(xq-xk+7)]
//               k==64      -> gem[q]
//               q==64,k<64 -> gem[k+1]
__global__ void __launch_bounds__(256) build_pos_kernel(
    const float* __restrict__ kg, const float* __restrict__ kf,
    const float* __restrict__ vg, const float* __restrict__ vf,
    uint16_t* __restrict__ kp, uint16_t* __restrict__ vp) {
  int idx = blockIdx.x * 256 + threadIdx.x;
  if (idx >= SEQ * SEQ) return;
  int q = idx / SEQ, k = idx - q * SEQ;
  const float *ks_, *vs_;
  if (k == 64)      { ks_ = kg + q * DK;       vs_ = vg + q * DK; }
  else if (q == 64) { ks_ = kg + (k + 1) * DK; vs_ = vg + (k + 1) * DK; }
  else {
    int yq = q >> 3, xq = q & 7, yk = k >> 3, xk = k & 7;
    int off = ((yq - yk + 7) * 15 + (xq - xk + 7)) * DK;
    ks_ = kf + off; vs_ = vf + off;
  }
  uint16_t* kd = kp + (size_t)idx * DK;
  uint16_t* vd = vp + (size_t)idx * DK;
  for (int d = 0; d < DK; ++d) { kd[d] = f2bf(ks_[d]); vd[d] = f2bf(vs_[d]); }
}

// ------------------------------------------------ unified bf16 WMMA GEMM
// C[r,e] = sum_d A[r,d] * W[e,d]   (A: ROWSxDM bf16, W: DMxDM bf16)
// grid (ROWS/128, DM/128), block 256 (8 waves).
// Workgroup tile: 128 rows x 128 cols; wave w: rows 16w..16w+15, 8 col-tiles.
// B k-slice (128x32 bf16 = 8 KB) double-buffered in LDS, filled by the CDNA5
// async global->LDS copy so the copy of step k+1 overlaps the WMMAs of step k.
// Epilogue: Cb != 0 -> bf16 store scaled by alpha;  else f32 store + residual.
__global__ void __launch_bounds__(256) gemm_kernel(
    const uint16_t* __restrict__ A, const uint16_t* __restrict__ W,
    uint16_t* __restrict__ Cb, float* __restrict__ Cf,
    const float* __restrict__ RES, float alpha) {
  __shared__ __align__(16) uint16_t sB[2][128 * 32];   // 2 x 8 KB

  const int t = threadIdx.x;
  const int wave = t >> 5, lane = t & 31;
  const int m = lane & 15, hh = lane >> 4;
  const int r0 = blockIdx.x * 128 + wave * 16;
  const int c0 = blockIdx.y * 128;

  v8f acc[8];
  #pragma unroll
  for (int n = 0; n < 8; ++n) acc[n] = zero8();

  // B staging map: thread t -> W row (c0 + t/2), 16-elem segment ((t&1)*16)
  const int brow = t >> 1, bseg = (t & 1) * 16;
  const uint16_t* aRow = A + (size_t)(r0 + m) * DM;

  auto stage = [&](int k0, int buf) {
    const uint16_t* wsrc = W + (size_t)(c0 + brow) * DM + k0 + bseg;
    uint16_t* ldst = &sB[buf][brow * 32 + bseg];
#ifdef USE_ASYNC_LDS
    async_g2l_b128(wsrc, ldst);
    async_g2l_b128(wsrc + 8, ldst + 8);
#else
    uint4 w0 = ((const uint4*)wsrc)[0], w1 = ((const uint4*)wsrc)[1];
    ((uint4*)ldst)[0] = w0; ((uint4*)ldst)[1] = w1;
#endif
  };

  stage(0, 0);                                        // prologue fill
  const int NSTEP = DM / 32;
  for (int ks = 0; ks < NSTEP; ++ks) {
    const int k0 = ks * 32, cur = ks & 1;
#ifdef USE_ASYNC_LDS
    __builtin_amdgcn_s_wait_asynccnt(0);              // my copies into buf[cur] done
#endif
    __syncthreads();                                  // everyone's done; prev reads done
    if (ks + 1 < NSTEP) stage(k0 + 32, cur ^ 1);      // overlap copy with compute

    Frag16 af;
    uint4 a0 = *(const uint4*)(aRow + k0 + hh * 16);
    uint4 a1 = *(const uint4*)(aRow + k0 + hh * 16 + 8);
    af.u[0] = a0.x; af.u[1] = a0.y; af.u[2] = a0.z; af.u[3] = a0.w;
    af.u[4] = a1.x; af.u[5] = a1.y; af.u[6] = a1.z; af.u[7] = a1.w;

    #pragma unroll
    for (int nt = 0; nt < 8; ++nt) {
      Frag16 bfr;
      const uint4* bp = (const uint4*)(&sB[cur][(nt * 16 + m) * 32 + hh * 16]);
      uint4 b0 = bp[0], b1 = bp[1];
      bfr.u[0] = b0.x; bfr.u[1] = b0.y; bfr.u[2] = b0.z; bfr.u[3] = b0.w;
      bfr.u[4] = b1.x; bfr.u[5] = b1.y; bfr.u[6] = b1.z; bfr.u[7] = b1.w;
      acc[nt] = wmma_bf16(af, bfr, acc[nt]);
    }
    gemm_sched_hint();            // enforce 8 x (2 ds_read -> 1 wmma) issue
  }

  if (Cb) {
    #pragma unroll
    for (int nt = 0; nt < 8; ++nt)
      #pragma unroll
      for (int r = 0; r < 8; ++r)
        Cb[(size_t)(r0 + r + 8 * hh) * DM + c0 + nt * 16 + m] = f2bf(alpha * acc[nt][r]);
  } else {
    #pragma unroll
    for (int nt = 0; nt < 8; ++nt)
      #pragma unroll
      for (int r = 0; r < 8; ++r) {
        size_t o = (size_t)(r0 + r + 8 * hh) * DM + c0 + nt * 16 + m;
        Cf[o] = acc[nt][r] + RES[o];
      }
  }
}

// ------------------------------------------------------- fused attention
// One workgroup per (b,h). Q already scaled by 1/sqrt(dk).
__global__ void __launch_bounds__(256) attn_kernel(
    const uint16_t* __restrict__ Q, const uint16_t* __restrict__ K,
    const uint16_t* __restrict__ V, const uint16_t* __restrict__ KP,
    const uint16_t* __restrict__ VP, uint16_t* __restrict__ AO,
    float* __restrict__ ATT) {
  // LDS plan (56,320 B): [sq 10240][sk 10240] -- reused as sout (20480 f32 B)
  //                      [svT 10240][slog 25600]
  __shared__ __align__(16) char smem[56320];
  uint16_t* sq  = (uint16_t*)smem;             // 80 x 64 bf16
  uint16_t* sk  = (uint16_t*)(smem + 10240);   // 80 x 64 bf16
  float*    sout= (float*)smem;                // 80 x 64 f32 (aliases sq+sk)
  uint16_t* svT = (uint16_t*)(smem + 20480);   // 64 x 80 bf16 (v transposed)
  float*    slog= (float*)(smem + 30720);      // 80 x 80 f32 logits/probs

  const int h = blockIdx.x, b = blockIdx.y;
  const int t = threadIdx.x;
  const int wave = t >> 5, lane = t & 31, m = lane & 15, hh = lane >> 4;
  const size_t rowBase = (size_t)b * (SEQ * DM) + (size_t)h * DK;

  for (int i = t; i < 64 * 80; i += 256) svT[i] = 0;
  __syncthreads();
  for (int i = t; i < 80 * 64; i += 256) {
    int s = i >> 6, d = i & 63;
    uint16_t qv = 0, kv = 0;
    if (s < SEQ) {
      size_t g = rowBase + (size_t)s * DM + d;
      qv = Q[g]; kv = K[g]; svT[d * 80 + s] = V[g];
    }
    sq[i] = qv; sk[i] = kv;
  }
  __syncthreads();

  // ---- logits = qs @ kh^T : 5x5 16x16 tiles, K = 64
  for (int j = wave; j < 25; j += 8) {
    int mt = j / 5, nt = j - mt * 5;
    v8f acc = zero8();
    #pragma unroll
    for (int k0 = 0; k0 < 64; k0 += 32) {
      Frag16 af, bfr;
      uint4 a0 = *(const uint4*)(sq + (mt * 16 + m) * 64 + k0 + hh * 16);
      uint4 a1 = *(const uint4*)(sq + (mt * 16 + m) * 64 + k0 + hh * 16 + 8);
      af.u[0] = a0.x; af.u[1] = a0.y; af.u[2] = a0.z; af.u[3] = a0.w;
      af.u[4] = a1.x; af.u[5] = a1.y; af.u[6] = a1.z; af.u[7] = a1.w;
      const uint4* bp = (const uint4*)(sk + (nt * 16 + m) * 64 + k0 + hh * 16);
      uint4 b0 = bp[0], b1 = bp[1];
      bfr.u[0] = b0.x; bfr.u[1] = b0.y; bfr.u[2] = b0.z; bfr.u[3] = b0.w;
      bfr.u[4] = b1.x; bfr.u[5] = b1.y; bfr.u[6] = b1.z; bfr.u[7] = b1.w;
      acc = wmma_bf16(af, bfr, acc);
    }
    #pragma unroll
    for (int r = 0; r < 8; ++r)
      slog[(mt * 16 + r + 8 * hh) * 80 + nt * 16 + m] = acc[r];
  }
  __syncthreads();

  // ---- positional bias: slog[q][k] += dot(qs[q,:], kpos[q,k,:])
  for (int idx = t; idx < SEQ * SEQ; idx += 256) {
    int q = idx / SEQ, k = idx - q * SEQ;
    const uint16_t* pp = KP + (size_t)idx * DK;
    const uint16_t* qp = sq + q * 64;
    float acc = 0.f;
    #pragma unroll 8
    for (int d = 0; d < 64; d += 2) {
      uint32_t pv = *(const uint32_t*)(pp + d);
      uint32_t qv = *(const uint32_t*)(qp + d);
      acc += bf2f(pv & 0xffffu) * bf2f(qv & 0xffffu);
      acc += bf2f(pv >> 16) * bf2f(qv >> 16);
    }
    slog[q * 80 + k] += acc;
  }
  __syncthreads();

  // ---- row softmax (65 entries), normalize in LDS
  if (t < SEQ) {
    float* row = slog + t * 80;
    float mx = row[0];
    for (int k = 1; k < SEQ; ++k) mx = fmaxf(mx, row[k]);
    float sum = 0.f;
    for (int k = 0; k < SEQ; ++k) { float e = __expf(row[k] - mx); row[k] = e; sum += e; }
    float inv = 1.f / sum;
    for (int k = 0; k < SEQ; ++k) row[k] *= inv;
  }
  __syncthreads();

  // ---- write fp32 probabilities to global attn output (all threads)
  {
    float* ao = ATT + (size_t)(b * NH + h) * (SEQ * SEQ);
    for (int i = t; i < SEQ * SEQ; i += 256) {
      int qq = i / SEQ, kk = i - qq * SEQ;
      ao[i] = slog[qq * 80 + kk];
    }
  }

  // ---- out = probs @ vh : 5x4 tiles, K = 0..63 WMMA + k=64 column scalarly
  for (int j = wave; j < 20; j += 8) {
    int mt = j >> 2, nt = j & 3;
    v8f acc = zero8();
    #pragma unroll
    for (int k0 = 0; k0 < 64; k0 += 32) {
      Frag16 af, bfr;
      const float* ar = slog + (mt * 16 + m) * 80;
      af.u[0] = pk2(ar[k0 + hh * 16 + 0], ar[k0 + hh * 16 + 1]);
      af.u[1] = pk2(ar[k0 + hh * 16 + 2], ar[k0 + hh * 16 + 3]);
      af.u[2] = pk2(ar[k0 + hh * 16 + 4], ar[k0 + hh * 16 + 5]);
      af.u[3] = pk2(ar[k0 + hh * 16 + 6], ar[k0 + hh * 16 + 7]);
      af.u[4] = pk2(ar[k0 + hh * 16 + 8], ar[k0 + hh * 16 + 9]);
      af.u[5] = pk2(ar[k0 + hh * 16 + 10], ar[k0 + hh * 16 + 11]);
      af.u[6] = pk2(ar[k0 + hh * 16 + 12], ar[k0 + hh * 16 + 13]);
      af.u[7] = pk2(ar[k0 + hh * 16 + 14], ar[k0 + hh * 16 + 15]);
      const uint4* bp = (const uint4*)(svT + (nt * 16 + m) * 80 + k0 + hh * 16);
      uint4 b0 = bp[0], b1 = bp[1];
      bfr.u[0] = b0.x; bfr.u[1] = b0.y; bfr.u[2] = b0.z; bfr.u[3] = b0.w;
      bfr.u[4] = b1.x; bfr.u[5] = b1.y; bfr.u[6] = b1.z; bfr.u[7] = b1.w;
      acc = wmma_bf16(af, bfr, acc);
    }
    float vv = bf2f((uint32_t)svT[(nt * 16 + m) * 80 + 64]);   // vh[64][col]
    #pragma unroll
    for (int r = 0; r < 8; ++r) {
      int row = mt * 16 + r + 8 * hh;
      acc[r] += slog[row * 80 + 64] * vv;                      // k = 64 term
      sout[row * 64 + nt * 16 + m] = acc[r];
    }
  }
  __syncthreads();

  // ---- positional V term: sout[q][d] += sum_k probs[q,k] * vpos[q,k,d]
  for (int q = t >> 2; q < SEQ; q += 64) {
    int dseg = (t & 3) * 16;
    float acc[16];
    #pragma unroll
    for (int d = 0; d < 16; ++d) acc[d] = 0.f;
    for (int k = 0; k < SEQ; ++k) {
      float a = slog[q * 80 + k];
      const uint16_t* vp = VP + ((size_t)(q * SEQ + k) * DK + dseg);
      #pragma unroll
      for (int d = 0; d < 16; d += 2) {
        uint32_t pv = *(const uint32_t*)(vp + d);
        acc[d]     += a * bf2f(pv & 0xffffu);
        acc[d + 1] += a * bf2f(pv >> 16);
      }
    }
    #pragma unroll
    for (int d = 0; d < 16; ++d) sout[q * 64 + dseg + d] += acc[d];
  }
  __syncthreads();

  // ---- store head output (bf16) back to (b*S, 1024) concat layout
  for (int i = t; i < SEQ * 64; i += 256) {
    int s = i >> 6, d = i & 63;
    AO[rowBase + (size_t)s * DM + d] = f2bf(sout[i]);
  }
}

// ---------------------------------------------------------------- launch
extern "C" void kernel_launch(void* const* d_in, const int* in_sizes, int n_in,
                              void* d_out, int out_size, void* d_ws, size_t ws_size,
                              hipStream_t stream) {
  (void)in_sizes; (void)n_in; (void)out_size; (void)ws_size;
  const float* q   = (const float*)d_in[0];
  const float* k   = (const float*)d_in[1];
  const float* v   = (const float*)d_in[2];
  const float* wqs = (const float*)d_in[3];
  const float* wks = (const float*)d_in[4];
  const float* wvs = (const float*)d_in[5];
  const float* wfc = (const float*)d_in[6];
  const float* kg  = (const float*)d_in[7];
  const float* kf  = (const float*)d_in[8];
  const float* vg  = (const float*)d_in[9];
  const float* vf  = (const float*)d_in[10];

  char* base = (char*)d_ws;
  size_t off = 0;
  auto carve = [&](size_t bytes) -> void* {
    void* p = base + off;
    off += (bytes + 255) & ~(size_t)255;
    return p;
  };
  uint16_t* wq_b = (uint16_t*)carve((size_t)DM * DM * 2);
  uint16_t* wk_b = (uint16_t*)carve((size_t)DM * DM * 2);
  uint16_t* wv_b = (uint16_t*)carve((size_t)DM * DM * 2);
  uint16_t* wf_b = (uint16_t*)carve((size_t)DM * DM * 2);
  uint16_t* qi_b = (uint16_t*)carve((size_t)ROWS * DM * 2);   // bf16 activations
  uint16_t* ki_b = (uint16_t*)carve((size_t)ROWS * DM * 2);
  uint16_t* vi_b = (uint16_t*)carve((size_t)ROWS * DM * 2);
  uint16_t* qs_b = (uint16_t*)carve((size_t)ROWS * DM * 2);   // projected heads
  uint16_t* kh_b = (uint16_t*)carve((size_t)ROWS * DM * 2);
  uint16_t* vh_b = (uint16_t*)carve((size_t)ROWS * DM * 2);
  uint16_t* ao_b = (uint16_t*)carve((size_t)ROWS * DM * 2);   // attention output
  uint16_t* kp_b = (uint16_t*)carve((size_t)SEQ * SEQ * DK * 2);
  uint16_t* vp_b = (uint16_t*)carve((size_t)SEQ * SEQ * DK * 2);

  float* out_p = (float*)d_out;
  float* att_p = out_p + OUT_ELEMS;

  const int NW4 = DM * DM / 4;
  const int NA4 = (int)(OUT_ELEMS / 4);
  cvt4_kernel<<<(NW4 + 255) / 256, 256, 0, stream>>>(wqs, wq_b, NW4);
  cvt4_kernel<<<(NW4 + 255) / 256, 256, 0, stream>>>(wks, wk_b, NW4);
  cvt4_kernel<<<(NW4 + 255) / 256, 256, 0, stream>>>(wvs, wv_b, NW4);
  cvt4_kernel<<<(NW4 + 255) / 256, 256, 0, stream>>>(wfc, wf_b, NW4);
  cvt4_kernel<<<(NA4 + 255) / 256, 256, 0, stream>>>(q, qi_b, NA4);
  cvt4_kernel<<<(NA4 + 255) / 256, 256, 0, stream>>>(k, ki_b, NA4);
  cvt4_kernel<<<(NA4 + 255) / 256, 256, 0, stream>>>(v, vi_b, NA4);

  build_pos_kernel<<<(SEQ * SEQ + 255) / 256, 256, 0, stream>>>(kg, kf, vg, vf, kp_b, vp_b);

  dim3 gg(ROWS / 128, DM / 128);
  gemm_kernel<<<gg, 256, 0, stream>>>(qi_b, wq_b, qs_b, nullptr, nullptr, 0.125f); // 1/sqrt(64)
  gemm_kernel<<<gg, 256, 0, stream>>>(ki_b, wk_b, kh_b, nullptr, nullptr, 1.0f);
  gemm_kernel<<<gg, 256, 0, stream>>>(vi_b, wv_b, vh_b, nullptr, nullptr, 1.0f);

  attn_kernel<<<dim3(NH, BATCH), 256, 0, stream>>>(qs_b, kh_b, vh_b, kp_b, vp_b, ao_b, att_p);

  gemm_kernel<<<gg, 256, 0, stream>>>(ao_b, wf_b, nullptr, out_p, q, 1.0f);
}
